// GPT2Attention_26980984554192
// MI455X (gfx1250) — compile-verified, matrix operand order checked by reading, not run on hardware
//
#include <hip/hip_runtime.h>

// ---------------------------------------------------------------------------
// GPT-2 attention block on gfx1250 via V_WMMA_F32_16X16X32_BF16.
// B=4, S=2048, D=1280, H=20, HD=64.
// ---------------------------------------------------------------------------

typedef __attribute__((ext_vector_type(16))) __bf16 v16bf;
typedef __attribute__((ext_vector_type(8)))  float  v8f;

struct __align__(16) U4 { unsigned int x, y, z, w; };

union AFrag {
  v16bf v;
  U4    q[2];
};

__device__ inline unsigned short f2bf(float f) {
  unsigned int u = __float_as_uint(f);
  u += 0x7FFFu + ((u >> 16) & 1u);          // round-to-nearest-even
  return (unsigned short)(u >> 16);
}

__device__ inline v8f wmma_bf16(const AFrag& a, const AFrag& b, v8f c) {
  return __builtin_amdgcn_wmma_f32_16x16x32_bf16(
      /*neg_a=*/false, a.v, /*neg_b=*/false, b.v,
      /*c_mod=*/(short)0, c, /*reuse_a=*/false, /*reuse_b=*/false);
}

// ---- async global->LDS staging (guarded: falls back to plain b128 copy) ----
#if defined(__has_builtin)
#if __has_builtin(__builtin_amdgcn_global_load_async_to_lds_b128) && \
    __has_builtin(__builtin_amdgcn_s_wait_asynccnt)
#define HAVE_ASYNC_LDS 1
#endif
#endif

#ifdef HAVE_ASYNC_LDS
typedef int vi4 __attribute__((__vector_size__(16)));
typedef vi4 __attribute__((address_space(1)))* gvi4p;   // global (AS1) int4*
typedef vi4 __attribute__((address_space(3)))* lvi4p;   // LDS    (AS3) int4*
#endif

__device__ inline void stage_b128(const void* g, void* l) {
#ifdef HAVE_ASYNC_LDS
  __builtin_amdgcn_global_load_async_to_lds_b128((gvi4p)g, (lvi4p)l, 0, 0);
#else
  *(U4*)l = *(const U4*)g;
#endif
}

__device__ inline void stage_wait() {
#ifdef HAVE_ASYNC_LDS
  __builtin_amdgcn_s_wait_asynccnt(0);
#endif
}

// ---------------------------------------------------------------------------
// Kernel 1: fused QKV projection. x[8192,1280] @ w[1280,3840] + bias,
// outputs bf16 Q/K/V in [B,H,S,HD] layout.
// Block = 256 threads (8 waves); block tile 256(M) x 64(N); K-step 32.
// Each wave computes a 32x64 strip (2 M-subtiles x 4 N-tiles = 8 WMMA/step).
// ---------------------------------------------------------------------------
__global__ __launch_bounds__(256) void qkv_gemm_kernel(
    const float* __restrict__ x, const float* __restrict__ w,
    const float* __restrict__ bias,
    unsigned short* __restrict__ Qo, unsigned short* __restrict__ Ko,
    unsigned short* __restrict__ Vo)
{
  __shared__ __align__(16) unsigned short lA[256 * 32];   // [m][k] bf16
  __shared__ __align__(16) unsigned short lB[64 * 32];    // [n][k] bf16 (w^T)

  const int t = threadIdx.x;
  const int wv = t >> 5, lane = t & 31, lhalf = lane >> 4, l16 = lane & 15;
  const int m0 = blockIdx.y * 256;
  const int n0 = blockIdx.x * 64;

  v8f zero = {};
  v8f acc[2][4] = {{zero, zero, zero, zero}, {zero, zero, zero, zero}};

  for (int k0 = 0; k0 < 1280; k0 += 32) {
    __syncthreads();
    { // stage x tile 256x32 f32 -> bf16 (float4 coalesced)
      int cg = t & 7, r0 = t >> 3;
      for (int it = 0; it < 8; ++it) {
        int row = r0 + it * 32;
        const float* src = x + (size_t)(m0 + row) * 1280 + k0 + cg * 4;
        float4 v4 = *(const float4*)src;
        unsigned short* dst = &lA[row * 32 + cg * 4];
        dst[0] = f2bf(v4.x); dst[1] = f2bf(v4.y);
        dst[2] = f2bf(v4.z); dst[3] = f2bf(v4.w);
      }
    }
    { // stage w^T tile: lB[n][k], reads coalesced along n
      int nn = t & 63, kb = t >> 6;
      for (int i = 0; i < 8; ++i) {
        int kk = kb * 8 + i;
        lB[nn * 32 + kk] = f2bf(w[(size_t)(k0 + kk) * 3840 + n0 + nn]);
        if (k0 + 32 < 1280)
          __builtin_prefetch(&w[(size_t)(k0 + 32 + kk) * 3840 + n0 + nn], 0, 1);
      }
    }
    __syncthreads();

    AFrag a[2];
    for (int mt = 0; mt < 2; ++mt) {
      const unsigned short* arow = &lA[(wv * 32 + mt * 16 + l16) * 32];
      a[mt].q[0] = *(const U4*)&arow[lhalf * 8];
      a[mt].q[1] = *(const U4*)&arow[16 + lhalf * 8];
    }
    for (int nt = 0; nt < 4; ++nt) {
      AFrag b;
      const unsigned short* brow = &lB[(nt * 16 + l16) * 32];
      b.q[0] = *(const U4*)&brow[lhalf * 8];
      b.q[1] = *(const U4*)&brow[16 + lhalf * 8];
      for (int mt = 0; mt < 2; ++mt)
        acc[mt][nt] = wmma_bf16(a[mt], b, acc[mt][nt]);
    }
  }

  // epilogue: bias add, scatter into Q/K/V [B,H,S,HD] bf16
  for (int nt = 0; nt < 4; ++nt) {
    int ng  = n0 + nt * 16 + l16;
    float bv = bias[ng];
    int sec = ng / 1280;
    int c   = ng - sec * 1280;
    int h = c >> 6, hd = c & 63;
    unsigned short* base = (sec == 0) ? Qo : (sec == 1) ? Ko : Vo;
    for (int mt = 0; mt < 2; ++mt) {
      for (int j = 0; j < 8; ++j) {
        int mg = m0 + wv * 32 + mt * 16 + lhalf * 8 + j;
        int bb = mg >> 11, s = mg & 2047;
        size_t dst = ((size_t)(bb * 20 + h) * 2048 + s) * 64 + hd;
        base[dst] = f2bf(acc[mt][nt][j] + bv);
      }
    }
  }
}

// ---------------------------------------------------------------------------
// Kernel 2: causal flash attention. One block per (b, h, 128 query rows).
// Each wave owns 16 query rows. KV tiles of 64, online softmax.
// ---------------------------------------------------------------------------
__global__ __launch_bounds__(256) void attn_kernel(
    const unsigned short* __restrict__ Q, const unsigned short* __restrict__ K,
    const unsigned short* __restrict__ V, unsigned short* __restrict__ O)
{
  __shared__ __align__(16) unsigned short lK [64 * 64];   // [s_k][d]
  __shared__ __align__(16) unsigned short lVT[64 * 64];   // [d][s_k]
  __shared__ __align__(16) float          lS[8][16 * 64]; // per-wave scores
  __shared__ __align__(16) unsigned short lP[8][16 * 64]; // per-wave probs
  __shared__ float lStat[8][16];

  const int t = threadIdx.x;
  const int wv = t >> 5, lane = t & 31, lhalf = lane >> 4, l16 = lane & 15;
  const int qt = blockIdx.x, h = blockIdx.y, b = blockIdx.z;
  const int qbase = qt * 128;
  const size_t headOff = (size_t)(b * 20 + h) * 2048 * 64;

  // Q fragments (16 rows x 64 d) held in registers
  AFrag qa[2];
  {
    const unsigned short* qrow = Q + headOff + (size_t)(qbase + wv * 16 + l16) * 64;
    for (int c = 0; c < 2; ++c) {
      qa[c].q[0] = *(const U4*)&qrow[c * 32 + lhalf * 8];
      qa[c].q[1] = *(const U4*)&qrow[c * 32 + 16 + lhalf * 8];
    }
  }

  v8f zero = {};
  v8f o[4] = {zero, zero, zero, zero};
  float m_run = -3.4e38f, l_run = 0.0f;

  const int nkv = qt * 2 + 2;                 // causal: only tiles <= q block
  for (int tk = 0; tk < nkv; ++tk) {
    const int kv0 = tk * 64;
    __syncthreads();
    { // stage K tile [s_k][d] (async b128) and V^T tile [d][s_k]
      int row = t >> 2, seg = t & 3;
      const unsigned short* ksrc = K + headOff + (size_t)(kv0 + row) * 64;
      stage_b128(&ksrc[seg * 16],     &lK[row * 64 + seg * 16]);
      stage_b128(&ksrc[seg * 16 + 8], &lK[row * 64 + seg * 16 + 8]);
      const unsigned short* vsrc = V + headOff + (size_t)(kv0 + row) * 64;
      for (int i = 0; i < 16; ++i) {
        int d = seg * 16 + i;
        lVT[d * 64 + row] = vsrc[d];
      }
      stage_wait();
    }
    __syncthreads();

    // scores = Q @ K^T  (contraction over d=64, two 32-chunks)
    v8f sf[4] = {zero, zero, zero, zero};
    for (int c = 0; c < 2; ++c)
      for (int nt = 0; nt < 4; ++nt) {
        AFrag kb;
        const unsigned short* krow = &lK[(nt * 16 + l16) * 64];
        kb.q[0] = *(const U4*)&krow[c * 32 + lhalf * 8];
        kb.q[1] = *(const U4*)&krow[c * 32 + 16 + lhalf * 8];
        sf[nt] = wmma_bf16(qa[c], kb, sf[nt]);
      }

    // scale + causal mask, spill to LDS for cross-lane softmax
    for (int nt = 0; nt < 4; ++nt) {
      int cglob = kv0 + nt * 16 + l16;
      for (int j = 0; j < 8; ++j) {
        int row = lhalf * 8 + j;
        int rglob = qbase + wv * 16 + row;
        float sv = sf[nt][j] * 0.125f;       // 1/sqrt(64)
        if (cglob > rglob) sv = -3.0e38f;
        lS[wv][row * 64 + nt * 16 + l16] = sv;
      }
    }
    __syncthreads();

    // online softmax: lane handles row l16, columns lhalf*32..+31
    {
      const float* srow = &lS[wv][l16 * 64 + lhalf * 32];
      float tmax = -3.4e38f;
      for (int i = 0; i < 32; ++i) tmax = fmaxf(tmax, srow[i]);
      tmax = fmaxf(tmax, __shfl_xor(tmax, 16, 32));
      float nmax = fmaxf(m_run, tmax);
      float corr = __expf(m_run - nmax);
      float psum = 0.0f;
      unsigned short* prow = &lP[wv][l16 * 64 + lhalf * 32];
      for (int i = 0; i < 32; ++i) {
        float p = __expf(srow[i] - nmax);
        psum += p;
        prow[i] = f2bf(p);
      }
      psum += __shfl_xor(psum, 16, 32);
      l_run = l_run * corr + psum;
      m_run = nmax;
      if (lhalf == 0) lStat[wv][l16] = corr;
    }
    __syncthreads();

    // rescale accumulators, then O += P @ V
    for (int j = 0; j < 8; ++j) {
      float cf = lStat[wv][lhalf * 8 + j];
      for (int nt = 0; nt < 4; ++nt) o[nt][j] *= cf;
    }
    for (int c = 0; c < 2; ++c) {
      AFrag pa;
      const unsigned short* prow = &lP[wv][l16 * 64];
      pa.q[0] = *(const U4*)&prow[c * 32 + lhalf * 8];
      pa.q[1] = *(const U4*)&prow[c * 32 + 16 + lhalf * 8];
      for (int nt = 0; nt < 4; ++nt) {
        AFrag vb;
        const unsigned short* vrow = &lVT[(nt * 16 + l16) * 64];
        vb.q[0] = *(const U4*)&vrow[c * 32 + lhalf * 8];
        vb.q[1] = *(const U4*)&vrow[c * 32 + 16 + lhalf * 8];
        o[nt] = wmma_bf16(pa, vb, o[nt]);
      }
    }
  }

  // finalize: divide by l, store bf16 into [B*S, D] for the proj GEMM
  if (lhalf == 0) lStat[wv][l16] = 1.0f / l_run;
  __syncthreads();
  for (int nt = 0; nt < 4; ++nt) {
    for (int j = 0; j < 8; ++j) {
      float inv = lStat[wv][lhalf * 8 + j];
      int row = qbase + wv * 16 + lhalf * 8 + j;
      size_t idx = ((size_t)(b * 2048 + row)) * 1280 + h * 64 + nt * 16 + l16;
      O[idx] = f2bf(o[nt][j] * inv);
    }
  }
}

// ---------------------------------------------------------------------------
// Kernel 3: output projection. A(bf16)[8192,1280] @ w_proj[1280,1280] + bias,
// f32 output. Block tile 256x64, per-wave 32x64.
// ---------------------------------------------------------------------------
__global__ __launch_bounds__(256) void proj_gemm_kernel(
    const unsigned short* __restrict__ A, const float* __restrict__ w,
    const float* __restrict__ bias, float* __restrict__ out)
{
  __shared__ __align__(16) unsigned short lA[256 * 32];
  __shared__ __align__(16) unsigned short lB[64 * 32];

  const int t = threadIdx.x;
  const int wv = t >> 5, lane = t & 31, lhalf = lane >> 4, l16 = lane & 15;
  const int m0 = blockIdx.y * 256;
  const int n0 = blockIdx.x * 64;

  v8f zero = {};
  v8f acc[2][4] = {{zero, zero, zero, zero}, {zero, zero, zero, zero}};

  for (int k0 = 0; k0 < 1280; k0 += 32) {
    __syncthreads();
    { // stage A tile (already bf16): 256 rows x 32 cols, async b128 copies
      int row = t >> 2, seg = t & 3;
      for (int it = 0; it < 4; ++it) {
        int r = row + it * 64;
        const unsigned short* src = A + (size_t)(m0 + r) * 1280 + k0 + seg * 8;
        stage_b128(src, &lA[r * 32 + seg * 8]);
      }
    }
    { // stage w^T tile
      int nn = t & 63, kb = t >> 6;
      for (int i = 0; i < 8; ++i) {
        int kk = kb * 8 + i;
        lB[nn * 32 + kk] = f2bf(w[(size_t)(k0 + kk) * 1280 + n0 + nn]);
        if (k0 + 32 < 1280)
          __builtin_prefetch(&w[(size_t)(k0 + 32 + kk) * 1280 + n0 + nn], 0, 1);
      }
    }
    stage_wait();
    __syncthreads();

    AFrag a[2];
    for (int mt = 0; mt < 2; ++mt) {
      const unsigned short* arow = &lA[(wv * 32 + mt * 16 + l16) * 32];
      a[mt].q[0] = *(const U4*)&arow[lhalf * 8];
      a[mt].q[1] = *(const U4*)&arow[16 + lhalf * 8];
    }
    for (int nt = 0; nt < 4; ++nt) {
      AFrag b;
      const unsigned short* brow = &lB[(nt * 16 + l16) * 32];
      b.q[0] = *(const U4*)&brow[lhalf * 8];
      b.q[1] = *(const U4*)&brow[16 + lhalf * 8];
      for (int mt = 0; mt < 2; ++mt)
        acc[mt][nt] = wmma_bf16(a[mt], b, acc[mt][nt]);
    }
  }

  for (int nt = 0; nt < 4; ++nt) {
    int ng = n0 + nt * 16 + l16;
    float bv = bias[ng];
    for (int mt = 0; mt < 2; ++mt) {
      for (int j = 0; j < 8; ++j) {
        int mg = m0 + wv * 32 + mt * 16 + lhalf * 8 + j;
        out[(size_t)mg * 1280 + ng] = acc[mt][nt][j] + bv;
      }
    }
  }
}

// ---------------------------------------------------------------------------
extern "C" void kernel_launch(void* const* d_in, const int* in_sizes, int n_in,
                              void* d_out, int out_size, void* d_ws, size_t ws_size,
                              hipStream_t stream) {
  (void)in_sizes; (void)n_in; (void)out_size; (void)ws_size;
  const float* x      = (const float*)d_in[0];
  const float* w_attn = (const float*)d_in[1];
  const float* b_attn = (const float*)d_in[2];
  const float* w_proj = (const float*)d_in[3];
  const float* b_proj = (const float*)d_in[4];
  float* out = (float*)d_out;

  unsigned short* ws = (unsigned short*)d_ws;
  const size_t HSZ = (size_t)4 * 20 * 2048 * 64;   // elements per Q/K/V buffer
  unsigned short* Qb = ws;
  unsigned short* Kb = ws + HSZ;
  unsigned short* Vb = ws + 2 * HSZ;
  unsigned short* Ab = ws + 3 * HSZ;               // attention output [B*S, D]

  qkv_gemm_kernel<<<dim3(60, 32), 256, 0, stream>>>(x, w_attn, b_attn, Qb, Kb, Vb);
  attn_kernel   <<<dim3(16, 20, 4), 256, 0, stream>>>(Qb, Kb, Vb, Ab);
  proj_gemm_kernel<<<dim3(20, 32), 256, 0, stream>>>(Ab, w_proj, b_proj, out);
}
